// LocalGNN_59227599012045
// MI455X (gfx1250) — matrix operand order repaired and checked by reading, not compiled, not run on hardware
//
#include <hip/hip_runtime.h>

#define DEV __device__ __forceinline__

typedef __attribute__((ext_vector_type(16))) __bf16 v16bf;
typedef __attribute__((ext_vector_type(8)))  float  v8f;

constexpr int Bv = 2048, Nv = 64, Dv = 128;
constexpr int BN = Bv * Nv;                 // 131072 rows of x
constexpr size_t BND = (size_t)BN * Dv;     // elements per [w] tensor

union AOp   { v16bf v; __bf16 e[16]; };
union Pack8 { uint4 q;  __bf16 e[8];  };

DEV v8f wmma_bf16(v16bf a, v16bf b, v8f c) {
  // D = A(16x32 bf16) * B(32x16 bf16) + C(16x16 f32)
  return __builtin_amdgcn_wmma_f32_16x16x32_bf16(false, a, false, b, (short)0, c,
                                                 false, false);
}

DEV float fsigmoid(float x) { return 1.f / (1.f + __expf(-x)); }
DEV float ftanh(float x) {
  x = fminf(fmaxf(x, -15.f), 15.f);
  float e = __expf(2.f * x);
  return (e - 1.f) / (e + 1.f);
}

// ---------------------------------------------------------------------------
// Kernel 0: pack weights transposed to bf16 (Wt[w][n][k] = W[k][n]) + biases.
// Order: 0..5 = ir,iz,im,xr,xz,xm (input-side), 6..8 = hr,hz,hm (recurrent).
// ---------------------------------------------------------------------------
__global__ __launch_bounds__(256) void k_prep(
    const float* W0, const float* W1, const float* W2, const float* W3,
    const float* W4, const float* W5, const float* W6, const float* W7,
    const float* W8,
    const float* c0, const float* c1, const float* c2, const float* c3,
    const float* c4, const float* c5, const float* c6, const float* c7,
    const float* c8,
    __bf16* __restrict__ Wt, float* __restrict__ bias) {
  const int w = blockIdx.x;
  const float* W = (w == 0) ? W0 : (w == 1) ? W1 : (w == 2) ? W2 : (w == 3) ? W3
                 : (w == 4) ? W4 : (w == 5) ? W5 : (w == 6) ? W6 : (w == 7) ? W7 : W8;
  const float* c = (w == 0) ? c0 : (w == 1) ? c1 : (w == 2) ? c2 : (w == 3) ? c3
                 : (w == 4) ? c4 : (w == 5) ? c5 : (w == 6) ? c6 : (w == 7) ? c7 : c8;
  for (int idx = threadIdx.x; idx < Dv * Dv; idx += 256) {
    const int n = idx >> 7, k = idx & 127;
    Wt[(size_t)w * Dv * Dv + idx] = (__bf16)W[k * Dv + n];
  }
  if (threadIdx.x < Dv) bias[w * Dv + threadIdx.x] = c[threadIdx.x];
}

// ---------------------------------------------------------------------------
// Kernel 1: six GEMMs out = x @ W + b over (BN,128)x(128,128).
// w<3 -> IT (Ir,Iz,Im), w>=3 -> AT init (Xr,Xz,Xm). Both stored TRANSPOSED:
// buf[w][b][d][t] bf16, so later B-operand / C-tile accesses are 16B vectors.
// Block = 256 thr (8 waves); wave owns 16 rows; grid = BN/128 = 1024.
// ---------------------------------------------------------------------------
__global__ __launch_bounds__(256) void k_proj(const float* __restrict__ x,
                                              const __bf16* __restrict__ Wt,
                                              const float* __restrict__ bias,
                                              __bf16* __restrict__ IT,
                                              __bf16* __restrict__ AT) {
  const int wave = threadIdx.x >> 5;
  const int lane = threadIdx.x & 31;
  const int hf   = lane >> 4;
  const int ln   = lane & 15;
  const int r0   = blockIdx.x * 128 + wave * 16;  // first of 16 rows
  const int b    = r0 >> 6;                       // batch index (16 | r0, 16|64)
  const int j0   = r0 & 63;                       // node index base

  // A-operands: 16x32 bf16 tiles of x, K = 0..127. A layout:
  // lane<16: K = k0..k0+15; lane>=16: K = k0+16..k0+31.
  AOp a[4];
  const float* xr = x + (size_t)(r0 + ln) * Dv;
#pragma unroll
  for (int ks = 0; ks < 4; ++ks) {
    const float* p1 = xr + ks * 32 + 16 * hf;
    const float* p2 = p1 + 8;
#pragma unroll
    for (int i = 0; i < 8; ++i) {
      a[ks].e[i]     = (__bf16)p1[i];
      a[ks].e[8 + i] = (__bf16)p2[i];
    }
  }

#pragma unroll 1
  for (int w = 0; w < 6; ++w) {
    const __bf16* Wb = Wt + (size_t)w * Dv * Dv;  // [n][k]
    const float*  bp = bias + w * Dv;
    __bf16* outp = (w < 3) ? (IT + (size_t)w * BND)
                           : (AT + (size_t)(w - 3) * BND);
#pragma unroll 1
    for (int nt = 0; nt < 8; ++nt) {
      const int n0 = nt * 16;
      v8f acc = {};
#pragma unroll
      for (int ks = 0; ks < 4; ++ks) {
        // B-operand: col n = n0+ln, K = ks*32 + e + 16*half (contiguous 16)
        const __bf16* wp = Wb + (size_t)(n0 + ln) * Dv + ks * 32 + 16 * hf;
        AOp bb;
#pragma unroll
        for (int i = 0; i < 16; ++i) bb.e[i] = wp[i];
        acc = wmma_bf16(a[ks].v, bb.v, acc);
      }
      const float bv = bp[n0 + ln];
      Pack8 s;
#pragma unroll
      for (int v = 0; v < 8; ++v) s.e[v] = (__bf16)(acc[v] + bv);
      // C-layout row m = v + 8*half -> j = j0 + v + 8*half : contiguous in v
      *(uint4*)(outp + (size_t)b * Dv * Nv + (size_t)(n0 + ln) * Nv + j0 + 8 * hf) = s.q;
    }
  }
}

// ---------------------------------------------------------------------------
// Kernel 2: AT[w][b] += g[b] @ I[w][b]   (N=64 rows, D=128 cols, K=64).
// One block per batch; 8 waves = (4 M-tiles) x (2 N-halves).
// Triangular K-skip: rows t0..t0+15 only need j <= t0+14.
// ---------------------------------------------------------------------------
__global__ __launch_bounds__(256) void k_graph(const float* __restrict__ graph,
                                               const __bf16* __restrict__ IT,
                                               __bf16* __restrict__ AT) {
  const int b    = blockIdx.x;
  const int wave = threadIdx.x >> 5;
  const int lane = threadIdx.x & 31;
  const int hf   = lane >> 4;
  const int ln   = lane & 15;
  const int t0   = (wave >> 1) * 16;
  const int nh   = wave & 1;
  const int nK   = (t0 + 46) >> 5;  // 1,1,2,2 K-steps for t0=0,16,32,48

  const float* gp = graph + (size_t)b * Nv * Nv + (size_t)(t0 + ln) * Nv;
  AOp ag0, ag1;
  {
    const float* p1 = gp + 8 * hf;
    const float* p2 = p1 + 16;
#pragma unroll
    for (int i = 0; i < 8; ++i) {
      const int j1 = 8 * hf + i, j2 = j1 + 16;
      ag0.e[i]     = (__bf16)((j1 == t0 + ln) ? 0.f : p1[i]);  // zero diagonal
      ag0.e[8 + i] = (__bf16)((j2 == t0 + ln) ? 0.f : p2[i]);
    }
  }
  if (nK == 2) {
    const float* p1 = gp + 32 + 8 * hf;
    const float* p2 = p1 + 16;
#pragma unroll
    for (int i = 0; i < 8; ++i) {
      const int j1 = 32 + 8 * hf + i, j2 = j1 + 16;
      ag1.e[i]     = (__bf16)((j1 == t0 + ln) ? 0.f : p1[i]);
      ag1.e[8 + i] = (__bf16)((j2 == t0 + ln) ? 0.f : p2[i]);
    }
  }

#pragma unroll 1
  for (int w = 0; w < 3; ++w) {
    const __bf16* Ib = IT + (size_t)w * BND + (size_t)b * Dv * Nv;  // [d][j]
    __bf16*       Ab = AT + (size_t)w * BND + (size_t)b * Dv * Nv;  // [d][t]
#pragma unroll 1
    for (int nt = 0; nt < 4; ++nt) {
      const int n0 = nh * 64 + nt * 16;
      __bf16* cp = Ab + (size_t)(n0 + ln) * Nv + t0 + 8 * hf;
      Pack8 s;
      s.q = *(const uint4*)cp;                 // accumulator init = X part
      v8f acc;
#pragma unroll
      for (int v = 0; v < 8; ++v) acc[v] = (float)s.e[v];
      {
        const __bf16* ip = Ib + (size_t)(n0 + ln) * Nv + 8 * hf;
        AOp bb;
#pragma unroll
        for (int i = 0; i < 16; ++i) bb.e[i] = ip[(i & 7) + ((i >> 3) << 4)];
        acc = wmma_bf16(ag0.v, bb.v, acc);
      }
      if (nK == 2) {
        const __bf16* ip = Ib + (size_t)(n0 + ln) * Nv + 32 + 8 * hf;
        AOp bb;
#pragma unroll
        for (int i = 0; i < 16; ++i) bb.e[i] = ip[(i & 7) + ((i >> 3) << 4)];
        acc = wmma_bf16(ag1.v, bb.v, acc);
      }
#pragma unroll
      for (int v = 0; v < 8; ++v) s.e[v] = (__bf16)acc[v];
      *(uint4*)cp = s.q;
    }
  }
}

// ---------------------------------------------------------------------------
// Kernel 3: GRU scan. One wave per block, 16 batch rows, t = 0..63.
// h kept in f32 C-layout registers; bf16 copy bounced through a 16x136 LDS
// tile to re-form the WMMA A-operand each step. Recurrent weights stream from
// global (96 KB working set, WGP$-resident). 128 blocks.
// ---------------------------------------------------------------------------
__global__ __launch_bounds__(32) void k_scan(const __bf16* __restrict__ AT,
                                             const __bf16* __restrict__ Wt,
                                             const float* __restrict__ bias,
                                             float* __restrict__ out) {
  __shared__ __bf16 hlds[16 * 136];   // 16 rows x 128 cols, +8 pad per row
  const int lane = threadIdx.x & 31;
  const int hf = lane >> 4, ln = lane & 15;
  const int b0 = blockIdx.x * 16;

  const __bf16* Whr = Wt + (size_t)6 * Dv * Dv;
  const __bf16* Whz = Wt + (size_t)7 * Dv * Dv;
  const __bf16* Whm = Wt + (size_t)8 * Dv * Dv;

  float bhr[8], bhz[8], bhm[8];
#pragma unroll
  for (int nt = 0; nt < 8; ++nt) {
    bhr[nt] = bias[6 * Dv + nt * 16 + ln];
    bhz[nt] = bias[7 * Dv + nt * 16 + ln];
    bhm[nt] = bias[8 * Dv + nt * 16 + ln];
  }

  float h[8][8];
#pragma unroll
  for (int nt = 0; nt < 8; ++nt)
#pragma unroll
    for (int v = 0; v < 8; ++v) h[nt][v] = 0.f;

  for (int i = threadIdx.x; i < 16 * 136; i += 32) hlds[i] = (__bf16)0.f;
  __syncthreads();

#pragma unroll 1
  for (int t = 0; t < Nv; ++t) {
    // h as A-operand (16x32 tiles, K = D)
    AOp ah[4];
#pragma unroll
    for (int ks = 0; ks < 4; ++ks) {
      const __bf16* p1 = &hlds[ln * 136 + ks * 32 + 16 * hf];
      const __bf16* p2 = p1 + 8;
#pragma unroll
      for (int i = 0; i < 8; ++i) { ah[ks].e[i] = p1[i]; ah[ks].e[8 + i] = p2[i]; }
    }

#pragma unroll 1
    for (int nt = 0; nt < 8; ++nt) {
      const int n0 = nt * 16;
      v8f ur = {}, uz = {}, um = {};
#pragma unroll
      for (int ks = 0; ks < 4; ++ks) {
        const size_t wo = (size_t)(n0 + ln) * Dv + ks * 32 + 16 * hf;
        AOp br, bz, bm;
#pragma unroll
        for (int i = 0; i < 16; ++i) {
          br.e[i] = Whr[wo + i]; bz.e[i] = Whz[wo + i]; bm.e[i] = Whm[wo + i];
        }
        ur = wmma_bf16(ah[ks].v, br.v, ur);
        uz = wmma_bf16(ah[ks].v, bz.v, uz);
        um = wmma_bf16(ah[ks].v, bm.v, um);
      }
#pragma unroll
      for (int v = 0; v < 8; ++v) {
        const int m = v + 8 * hf;
        const size_t base =
            (size_t)(b0 + m) * (Dv * Nv) + (size_t)(n0 + ln) * Nv + t;
        const float ar = (float)AT[0 * BND + base];
        const float az = (float)AT[1 * BND + base];
        const float am = (float)AT[2 * BND + base];
        const float r  = fsigmoid(ar + ur[v] + bhr[nt]);
        const float z  = fsigmoid(az + uz[v] + bhz[nt]);
        const float mm = ftanh(am + r * (um[v] + bhm[nt]));
        const float hn = (1.f - z) * mm + z * h[nt][v];
        h[nt][v] = hn;
        out[(size_t)(b0 + m) * (Nv * Dv) + (size_t)t * Dv + n0 + ln] =
            fmaxf(hn, 0.f);
        hlds[m * 136 + n0 + ln] = (__bf16)hn;  // next step's A-operand
      }
    }
  }
}

// ---------------------------------------------------------------------------
extern "C" void kernel_launch(void* const* d_in, const int* in_sizes, int n_in,
                              void* d_out, int out_size, void* d_ws, size_t ws_size,
                              hipStream_t stream) {
  (void)in_sizes; (void)n_in; (void)out_size; (void)ws_size;
  const float* x     = (const float*)d_in[0];
  const float* graph = (const float*)d_in[1];
  const float* W_ir = (const float*)d_in[2];  const float* b_ir = (const float*)d_in[3];
  const float* W_hr = (const float*)d_in[4];  const float* b_hr = (const float*)d_in[5];
  const float* W_xr = (const float*)d_in[6];  const float* b_xr = (const float*)d_in[7];
  const float* W_iz = (const float*)d_in[8];  const float* b_iz = (const float*)d_in[9];
  const float* W_hz = (const float*)d_in[10]; const float* b_hz = (const float*)d_in[11];
  const float* W_xz = (const float*)d_in[12]; const float* b_xz = (const float*)d_in[13];
  const float* W_im = (const float*)d_in[14]; const float* b_im = (const float*)d_in[15];
  const float* W_hm = (const float*)d_in[16]; const float* b_hm = (const float*)d_in[17];
  const float* W_xm = (const float*)d_in[18]; const float* b_xm = (const float*)d_in[19];

  char* ws = (char*)d_ws;
  const size_t off_bias = (size_t)9 * 128 * 128 * 2;          // 294912
  const size_t off_IT   = 299520;                              // 512-aligned
  const size_t off_AT   = off_IT + 2 * (size_t)3 * BND;        // +100663296
  __bf16* Wt   = (__bf16*)(ws);
  float*  bias = (float*)(ws + off_bias);
  __bf16* IT   = (__bf16*)(ws + off_IT);
  __bf16* AT   = (__bf16*)(ws + off_AT);

  k_prep<<<9, 256, 0, stream>>>(W_ir, W_iz, W_im, W_xr, W_xz, W_xm, W_hr, W_hz, W_hm,
                                b_ir, b_iz, b_im, b_xr, b_xz, b_xm, b_hr, b_hz, b_hm,
                                Wt, bias);
  k_proj<<<BN / 128, 256, 0, stream>>>(x, Wt, bias, IT, AT);
  k_graph<<<Bv, 256, 0, stream>>>(graph, IT, AT);
  k_scan<<<Bv / 16, 32, 0, stream>>>(AT, Wt, bias, (float*)d_out);
}
